// GNN_74259984548159
// MI455X (gfx1250) — compile-verified, matrix operand order checked by reading, not run on hardware
//
#include <hip/hip_runtime.h>

// GraphSAGE 2-layer forward for MI455X (gfx1250, wave32).
// Edge aggregation: f32 global atomics (L2-resident accumulators).
// Dense layers: v_wmma_f32_16x16x32_bf16 with f32 accumulate.

#define D_FEAT 64
#define HIDDEN 128
#define OUTD   16

typedef __attribute__((ext_vector_type(16))) __bf16 v16bf;
typedef __attribute__((ext_vector_type(8)))  float  v8f;

union Frag { v16bf v; unsigned u[8]; };

__device__ __forceinline__ unsigned short f2bf(float f) {
    unsigned u = __float_as_uint(f);
    u += 0x7FFFu + ((u >> 16) & 1u);          // round-to-nearest-even
    return (unsigned short)(u >> 16);
}

// K-index base within a 32-wide K chunk for 16-bit A/B fragments
// (ISA 7.12.2: VGPR v<4 holds K=2v+8*hi, v>=4 holds K=2v+8+8*hi).
__device__ __forceinline__ int kb16(int v, int hi) {
    return (v < 4) ? (2 * v + 8 * hi) : (2 * v + 8 + 8 * hi);
}

// ---------------------------------------------------------------- zero
__global__ void k_zero(float* p, long n) {
    long i = (long)blockIdx.x * blockDim.x + threadIdx.x;
    long s = (long)gridDim.x * blockDim.x;
    for (; i < n; i += s) p[i] = 0.0f;
}

// ---------------------------------------------------------------- degree
__global__ void k_deg(const int* dst, float* deg, int E) {
    int i = blockIdx.x * blockDim.x + threadIdx.x;
    if (i < E) atomicAdd(&deg[dst[i]], 1.0f);
}

// ---------------------------------------------------------------- scatter layer1 (64 feats, 2/lane)
__global__ void k_scatter1(const float* __restrict__ x, const int* src, const int* dst,
                           float* agg1, int E) {
    long gid = (long)blockIdx.x * blockDim.x + threadIdx.x;
    int e = (int)(gid >> 5);
    if (e >= E) return;
    int f = ((int)gid & 31) * 2;
    float2 v = *(const float2*)(x + (long)src[e] * D_FEAT + f);
    float* a = agg1 + (long)dst[e] * D_FEAT + f;
    atomicAdd(a,     v.x);
    atomicAdd(a + 1, v.y);
}

// ---------------------------------------------------------------- scatter layer2 (128 feats, 4/lane)
__global__ void k_scatter2(const float* __restrict__ h, const int* src, const int* dst,
                           float* agg2, int E) {
    long gid = (long)blockIdx.x * blockDim.x + threadIdx.x;
    int e = (int)(gid >> 5);
    if (e >= E) return;
    int f = ((int)gid & 31) * 4;
    float4 v = *(const float4*)(h + (long)src[e] * HIDDEN + f);
    float* a = agg2 + (long)dst[e] * HIDDEN + f;
    atomicAdd(a,     v.x);
    atomicAdd(a + 1, v.y);
    atomicAdd(a + 2, v.z);
    atomicAdd(a + 3, v.w);
}

// ------------------------------------------------- pack B1 = [W1_l;W1_r] (128x128) -> bf16 frags
// layout: [t(8)][kc(4)][lane(32)][16 halfs]  (per-lane contiguous 32B)
__global__ void k_pack_w1(const float* W1l, const float* W1r, unsigned short* Bp) {
    int tid = blockIdx.x * blockDim.x + threadIdx.x;
    if (tid >= 8 * 4 * 32 * 16) return;
    int idx = tid & 15, lane = (tid >> 4) & 31, kc = (tid >> 9) & 3, t = tid >> 11;
    int lo = lane & 15, hi = lane >> 4, v = idx >> 1, j = idx & 1;
    int k = kc * 32 + kb16(v, hi) + j;
    int n = t * 16 + lo;
    float val = (k < 64) ? W1l[k * HIDDEN + n] : W1r[(k - 64) * HIDDEN + n];
    Bp[tid] = f2bf(val);
}

// ------------------------------------------------- pack B2 = [W2_l;W2_r] (256x16) -> bf16 frags
// layout: [kc(8)][lane(32)][16 halfs]
__global__ void k_pack_w2(const float* W2l, const float* W2r, unsigned short* Bp) {
    int tid = blockIdx.x * blockDim.x + threadIdx.x;
    if (tid >= 8 * 32 * 16) return;
    int idx = tid & 15, lane = (tid >> 4) & 31, kc = tid >> 9;
    int lo = lane & 15, hi = lane >> 4, v = idx >> 1, j = idx & 1;
    int k = kc * 32 + kb16(v, hi) + j;
    float val = (k < 128) ? W2l[k * OUTD + lo] : W2r[(k - 128) * OUTD + lo];
    Bp[tid] = f2bf(val);
}

// ------------------------------------------------- GEMM1: h = relu([mean|x] @ B1 + b1)
// block = 256 thr (8 waves), one 16-node M-tile per block, wave w owns N-cols [16w,16w+16)
__global__ void __launch_bounds__(256) k_gemm1(const float* __restrict__ x,
                                               const float* __restrict__ agg1,
                                               const float* __restrict__ deg,
                                               const unsigned short* __restrict__ Bp,
                                               const float* __restrict__ b1,
                                               float* __restrict__ h, int nN) {
    __shared__ __align__(16) unsigned short As[16 * 128];   // [m][k] bf16, K = 128 = [mean|x]
    int base = blockIdx.x * 16;
    int tid  = threadIdx.x;
    #pragma unroll
    for (int i = 0; i < 8; ++i) {
        int e = tid + i * 256;
        int m = e >> 7, k = e & 127;
        int node = base + m;
        float val = 0.0f;
        if (node < nN) {
            if (k < 64) val = agg1[(long)node * 64 + k] / fmaxf(deg[node], 1.0f);
            else        val = x[(long)node * 64 + (k - 64)];
        }
        As[e] = f2bf(val);
    }
    __syncthreads();

    int wave = tid >> 5, lane = tid & 31;
    int lo = lane & 15, hi = lane >> 4;
    const unsigned* As32 = (const unsigned*)As;
    v8f c = {};
    for (int kc = 0; kc < 4; ++kc) {
        Frag a, b;
        #pragma unroll
        for (int v = 0; v < 8; ++v) {
            int kb = kc * 32 + kb16(v, hi);
            a.u[v] = As32[lo * 64 + (kb >> 1)];
        }
        const unsigned* q = (const unsigned*)(Bp + (((wave * 4 + kc) * 32 + lane) << 4));
        #pragma unroll
        for (int v = 0; v < 8; ++v) b.u[v] = q[v];
        c = __builtin_amdgcn_wmma_f32_16x16x32_bf16(false, a.v, false, b.v,
                                                    (short)0, c, false, false);
    }
    int n = wave * 16 + lo;
    float bias = b1[n];
    #pragma unroll
    for (int v = 0; v < 8; ++v) {
        int node = base + v + 8 * hi;
        if (node < nN) h[(long)node * HIDDEN + n] = fmaxf(c[v] + bias, 0.0f);
    }
}

// ------------------------------------------------- GEMM2 + softmax
// block = 128 thr (4 waves), 64 nodes/block, wave w owns its own 16-node tile, N = 16 (one tile)
__global__ void __launch_bounds__(128) k_gemm2(const float* __restrict__ hbuf,
                                               const float* __restrict__ agg2,
                                               const float* __restrict__ deg,
                                               const unsigned short* __restrict__ Bp,
                                               const float* __restrict__ b2,
                                               float* __restrict__ outp, int nN) {
    __shared__ __align__(16) unsigned short As[64 * 256];   // [m][k] bf16, K = 256 = [mean2|h]
    int base = blockIdx.x * 64;
    int tid  = threadIdx.x;
    #pragma unroll 4
    for (int i = 0; i < 128; ++i) {
        int e = tid + i * 128;
        int m = e >> 8, k = e & 255;
        int node = base + m;
        float val = 0.0f;
        if (node < nN) {
            if (k < 128) val = agg2[(long)node * HIDDEN + k] / fmaxf(deg[node], 1.0f);
            else         val = hbuf[(long)node * HIDDEN + (k - 128)];
        }
        As[e] = f2bf(val);
    }
    __syncthreads();

    int wave = tid >> 5, lane = tid & 31;
    int lo = lane & 15, hi = lane >> 4;
    const unsigned* As32 = (const unsigned*)As;
    v8f c = {};
    for (int kc = 0; kc < 8; ++kc) {
        Frag a, b;
        #pragma unroll
        for (int v = 0; v < 8; ++v) {
            int kb = kc * 32 + kb16(v, hi);
            a.u[v] = As32[(wave * 16 + lo) * 128 + (kb >> 1)];
        }
        const unsigned* q = (const unsigned*)(Bp + ((kc * 32 + lane) << 4));
        #pragma unroll
        for (int v = 0; v < 8; ++v) b.u[v] = q[v];
        c = __builtin_amdgcn_wmma_f32_16x16x32_bf16(false, a.v, false, b.v,
                                                    (short)0, c, false, false);
    }
    float bias = b2[lo];
    #pragma unroll
    for (int v = 0; v < 8; ++v) {
        float d = c[v] + bias;
        float mx = d;                                  // row lives across 16 lanes (lo)
        #pragma unroll
        for (int off = 8; off >= 1; off >>= 1) mx = fmaxf(mx, __shfl_xor(mx, off, 16));
        float ex = __expf(d - mx);
        float sm = ex;
        #pragma unroll
        for (int off = 8; off >= 1; off >>= 1) sm += __shfl_xor(sm, off, 16);
        int node = base + wave * 16 + v + 8 * hi;
        if (node < nN) outp[(long)node * OUTD + lo] = ex / sm;
    }
}

// ----------------------------------------------------------------------
extern "C" void kernel_launch(void* const* d_in, const int* in_sizes, int n_in,
                              void* d_out, int out_size, void* d_ws, size_t ws_size,
                              hipStream_t stream) {
    const float* x   = (const float*)d_in[0];
    const int*   ei  = (const int*)  d_in[1];
    const float* W1l = (const float*)d_in[2];
    const float* W1r = (const float*)d_in[3];
    const float* b1  = (const float*)d_in[4];
    const float* W2l = (const float*)d_in[5];
    const float* W2r = (const float*)d_in[6];
    const float* b2  = (const float*)d_in[7];
    float* out = (float*)d_out;

    int N = in_sizes[0] / D_FEAT;     // 50000
    int E = in_sizes[1] / 2;          // 800000
    const int* src = ei;
    const int* dst = ei + E;

    // workspace layout (floats): deg | agg1 | agg2 | h | Bpack1(u16) | Bpack2(u16)
    float* W    = (float*)d_ws;
    float* deg  = W;                                  // N
    float* agg1 = W + N;                              // N*64
    float* agg2 = agg1 + (long)N * 64;                // N*128
    float* hbuf = agg2 + (long)N * 128;               // N*128
    long  packOff = (((long)N * (1 + 64 + 128 + 128)) + 15) & ~15L;   // 64B aligned
    unsigned short* Bp1 = (unsigned short*)(W + packOff);             // 16384 u16
    unsigned short* Bp2 = Bp1 + 16384;                                // 4096 u16

    long zn = (long)N * (1 + 64 + 128);               // deg + agg1 + agg2
    k_zero<<<2048, 256, 0, stream>>>(W, zn);

    k_pack_w1<<<(8 * 4 * 32 * 16 + 255) / 256, 256, 0, stream>>>(W1l, W1r, Bp1);
    k_pack_w2<<<(8 * 32 * 16 + 255) / 256, 256, 0, stream>>>(W2l, W2r, Bp2);

    k_deg<<<(E + 255) / 256, 256, 0, stream>>>(dst, deg, E);

    long t1 = (long)E * 32;
    k_scatter1<<<(int)((t1 + 255) / 256), 256, 0, stream>>>(x, src, dst, agg1, E);

    k_gemm1<<<(N + 15) / 16, 256, 0, stream>>>(x, agg1, deg, Bp1, b1, hbuf, N);

    k_scatter2<<<(int)((t1 + 255) / 256), 256, 0, stream>>>(hbuf, src, dst, agg2, E);

    k_gemm2<<<(N + 63) / 64, 128, 0, stream>>>(hbuf, agg2, deg, Bp2, b2, out, N);
}
